// mepognn_1185410974018
// MI455X (gfx1250) — compile-verified, matrix operand order checked by reading, not run on hardware
//
#include <hip/hip_runtime.h>
#include <hip/hip_bf16.h>
#include <math.h>

// ---------------------------------------------------------------------------
// MEPO-GNN forward on MI455X (gfx1250), wave32 + WMMA bf16.
// Graph matmuls [A1;A2] @ X (~192 GFLOP) + end 1x1 convs (~50 GFLOP) via
// v_wmma_f32_16x16x32_bf16. B operands stored transposed (NxK); LDS staging
// uses CDNA5 global_load_async_to_lds_b128 (ASYNCcnt) with double-buffered
// tiles so the DMA of tile k+1 overlaps the WMMAs of tile k.
// ---------------------------------------------------------------------------

typedef __bf16 bf16_t;
typedef __attribute__((ext_vector_type(16))) __bf16 v16bf;
typedef __attribute__((ext_vector_type(8)))  __bf16 v8bf;
typedef __attribute__((ext_vector_type(8)))  float  v8f;

#define V_N   2000
#define KP    2016           // V padded to multiple of 32 for WMMA K loop
#define MP12  4096           // A12 rows padded to multiple of 128 (branch-free staging)
#define BATCH 8
#define RC    32
#define DC    32
#define SC    256
#define EC    512
#define NLAY  6
#define NBV   16000          // BATCH * V_N
#define CDIV(a,b) (((a)+(b)-1)/(b))

__device__ __forceinline__ float sigf(float x) { return 1.0f / (1.0f + expf(-x)); }

// ---------------- mobility / adjacency ----------------
__global__ void k_mob_build(const float* __restrict__ g, const float* __restrict__ maxlog,
                            float* __restrict__ mob, float* __restrict__ mobT) {
  int idx = blockIdx.x * blockDim.x + threadIdx.x;
  if (idx >= V_N * V_N) return;
  int v = idx / V_N, w = idx % V_N;
  float m = expf(fmaxf(g[idx] * maxlog[0], 0.0f));
  mob[idx] = m;
  mobT[(size_t)w * V_N + v] = m;
}

__global__ void k_rowsum(const float* __restrict__ mob, const float* __restrict__ mobT,
                         float* __restrict__ rs) {
  __shared__ float red[256];
  int row = blockIdx.x;
  const float* src = (row < V_N) ? (mob + (size_t)row * V_N) : (mobT + (size_t)(row - V_N) * V_N);
  float s = 0.0f;
  for (int j = threadIdx.x; j < V_N; j += 256) s += src[j];
  red[threadIdx.x] = s;
  __syncthreads();
  for (int st = 128; st > 0; st >>= 1) {
    if (threadIdx.x < st) red[threadIdx.x] += red[threadIdx.x + st];
    __syncthreads();
  }
  if (threadIdx.x == 0) rs[row] = red[0];
}

// A12: rows 0..1999 = A1 (row-normalized mob), rows 2000..3999 = A2
// (row-normalized mobT). Rows 4000..4095 and cols >= V_N zero-filled.
__global__ void k_a12(const float* __restrict__ mob, const float* __restrict__ mobT,
                      const float* __restrict__ rs, bf16_t* __restrict__ A12) {
  int idx = blockIdx.x * blockDim.x + threadIdx.x;
  if (idx >= MP12 * KP) return;
  int r = idx / KP, k = idx % KP;
  float v = 0.0f;
  if (k < V_N && r < 2 * V_N) {
    float m = (r < V_N) ? mob[(size_t)r * V_N + k] : mobT[(size_t)(r - V_N) * V_N + k];
    v = m / rs[r];
  }
  A12[idx] = (bf16_t)v;
}

// ---------------- start conv (1x1, with left time pad) ----------------
__global__ void k_start(const float* __restrict__ xn, const float* __restrict__ w,
                        const float* __restrict__ b, float* __restrict__ x) {
  int idx = blockIdx.x * blockDim.x + threadIdx.x;
  const int T = 15;
  if (idx >= BATCH * RC * V_N * T) return;
  int l = idx % T;
  int v = (idx / T) % V_N;
  int c = (idx / (T * V_N)) % RC;
  int bb = idx / (T * V_N * RC);
  float acc = b[c];
  if (l > 0) {
    for (int i = 0; i < 4; ++i)
      acc += w[c * 4 + i] * xn[((size_t)(bb * 4 + i) * V_N + v) * 14 + (l - 1)];
  }
  x[idx] = acc;
}

// ---------------- dilated filter/gate conv, fused tanh*sigmoid ----------------
__global__ __launch_bounds__(256)
void k_filtgate(const float* __restrict__ x, const float* __restrict__ fw,
                const float* __restrict__ fb, const float* __restrict__ gw,
                const float* __restrict__ gb, float* __restrict__ x1,
                int Tin, int Tout, int dil) {
  __shared__ float sfw[2048], sgw[2048], sfb[32], sgb[32];
  for (int j = threadIdx.x; j < 2048; j += 256) { sfw[j] = fw[j]; sgw[j] = gw[j]; }
  if (threadIdx.x < 32) { sfb[threadIdx.x] = fb[threadIdx.x]; sgb[threadIdx.x] = gb[threadIdx.x]; }
  __syncthreads();
  int idx = blockIdx.x * blockDim.x + threadIdx.x;
  int total = BATCH * DC * V_N * Tout;
  if (idx >= total) return;
  int l = idx % Tout;
  int v = (idx / Tout) % V_N;
  int co = (idx / (Tout * V_N)) % DC;
  int bb = idx / (Tout * V_N * DC);
  float af = sfb[co], ag = sgb[co];
  for (int ci = 0; ci < RC; ++ci) {
    size_t bi = ((size_t)(bb * RC + ci) * V_N + v) * Tin + l;
    float x0 = x[bi], xd = x[bi + dil];
    int wb = (co * RC + ci) * 2;
    af += sfw[wb] * x0 + sfw[wb + 1] * xd;
    ag += sgw[wb] * x0 + sgw[wb + 1] * xd;
  }
  x1[idx] = tanhf(af) * sigf(ag);
}

// ---------------- skip conv: only the LAST time slice matters (Tf==1) -------
// feat stored transposed: (NBV x 1536) so it is B^T for the end GEMM.
__global__ void k_skip(const float* __restrict__ x1, const float* __restrict__ sw,
                       const float* __restrict__ sb, bf16_t* __restrict__ feat,
                       int Tout, int chanBase) {
  int idx = blockIdx.x * blockDim.x + threadIdx.x;
  if (idx >= BATCH * SC * V_N) return;
  int v = idx % V_N;
  int sc = (idx / V_N) % SC;
  int bb = idx / (V_N * SC);
  float acc = sb[sc];
  for (int ci = 0; ci < DC; ++ci)
    acc += sw[sc * DC + ci] * x1[((size_t)(bb * DC + ci) * V_N + v) * Tout + (Tout - 1)];
  feat[(size_t)(bb * V_N + v) * 1536 + chanBase + sc] = (bf16_t)fmaxf(acc, 0.0f);
}

// ---------------- pack x1 -> bf16 X^T (N x KP) for graph GEMM ----------------
__global__ void k_pack_x(const float* __restrict__ x1, bf16_t* __restrict__ Xt, int Tout) {
  int N = BATCH * DC * Tout;
  int idx = blockIdx.x * blockDim.x + threadIdx.x;
  if (idx >= N * KP) return;
  int n = idx / KP, k = idx % KP;
  float v = 0.0f;
  if (k < V_N) {
    int bb = n / (DC * Tout);
    int r = n % (DC * Tout);
    int c = r / Tout, l = r % Tout;
    v = x1[((size_t)(bb * DC + c) * V_N + k) * Tout + l];
  }
  Xt[idx] = (bf16_t)v;
}

// ---------------- bf16 WMMA GEMM: C[M,N] f32 = A[M,K] * BT[N,K]^T ----------------
// K multiple of 32, N multiple of 64. A must be padded to ceil(M/128)*128
// rows (loads are branch-free); C stores guarded by M. 256 threads = 8
// waves, block tile 128(M) x 64(N); each wave owns 32x32 (2x2 WMMA frags).
// Staging: double-buffered global_load_async_to_lds_b128 (ASYNCcnt).
__global__ __launch_bounds__(256)
void k_gemm_bf16(const bf16_t* __restrict__ A, const bf16_t* __restrict__ BT,
                 float* __restrict__ C, int M, int N, int K) {
  __shared__ alignas(16) bf16_t As[2][128][32];
  __shared__ alignas(16) bf16_t Bs[2][64][32];
  int tid = threadIdx.x;
  int wave = tid >> 5, lane = tid & 31;
  int half = lane >> 4, lm = lane & 15;
  int m0 = blockIdx.x * 128;
  int n0 = blockIdx.y * 64;
  int mw = (wave & 3) * 32;       // wave M offset in tile
  int nw = (wave >> 2) * 32;      // wave N offset in tile
  v8f acc00 = {0.f,0.f,0.f,0.f,0.f,0.f,0.f,0.f};
  v8f acc01 = acc00, acc10 = acc00, acc11 = acc00;

  int arow = tid >> 1, acol = (tid & 1) * 16;   // A: 16 bf16 per thread (2 x b128)
  int brow = tid >> 2, bcol = (tid & 3) * 8;    // B: 8 bf16 per thread (1 x b128)

  const bf16_t* ga = A + (size_t)(m0 + arow) * K + acol;
  const bf16_t* gb = BT + (size_t)(n0 + brow) * K + bcol;
  unsigned ldsA0 = (unsigned)(uintptr_t)&As[0][arow][acol];
  unsigned ldsA1 = (unsigned)(uintptr_t)&As[1][arow][acol];
  unsigned ldsB0 = (unsigned)(uintptr_t)&Bs[0][brow][bcol];
  unsigned ldsB1 = (unsigned)(uintptr_t)&Bs[1][brow][bcol];

  // prologue: stage tile 0 into buffer 0 (async DMA, ASYNCcnt += 3)
  asm volatile(
      "global_load_async_to_lds_b128 %0, %2, off\n\t"
      "global_load_async_to_lds_b128 %0, %2, off offset:16\n\t"
      "global_load_async_to_lds_b128 %1, %3, off"
      :: "v"(ldsA0), "v"(ldsB0), "v"(ga), "v"(gb) : "memory");

  int buf = 0;
  for (int k0 = 0; k0 < K; k0 += 32) {
    if (k0 + 32 < K) {
      const bf16_t* gan = ga + k0 + 32;
      const bf16_t* gbn = gb + k0 + 32;
      unsigned la = buf ? ldsA0 : ldsA1;
      unsigned lb = buf ? ldsB0 : ldsB1;
      asm volatile(
          "global_load_async_to_lds_b128 %0, %2, off\n\t"
          "global_load_async_to_lds_b128 %0, %2, off offset:16\n\t"
          "global_load_async_to_lds_b128 %1, %3, off"
          :: "v"(la), "v"(lb), "v"(gan), "v"(gbn) : "memory");
      // current tile's 3 loads complete when <=3 remain (in-order completion)
      asm volatile("s_wait_asynccnt 0x3" ::: "memory");
    } else {
      asm volatile("s_wait_asynccnt 0x0" ::: "memory");
    }
    __syncthreads();

    const v8bf* pa0 = reinterpret_cast<const v8bf*>(&As[buf][mw + lm][0]);
    const v8bf* pa1 = reinterpret_cast<const v8bf*>(&As[buf][mw + 16 + lm][0]);
    const v8bf* pb0 = reinterpret_cast<const v8bf*>(&Bs[buf][nw + lm][0]);
    const v8bf* pb1 = reinterpret_cast<const v8bf*>(&Bs[buf][nw + 16 + lm][0]);
    v16bf a0 = __builtin_shufflevector(pa0[half], pa0[2 + half],
                                       0,1,2,3,4,5,6,7,8,9,10,11,12,13,14,15);
    v16bf a1 = __builtin_shufflevector(pa1[half], pa1[2 + half],
                                       0,1,2,3,4,5,6,7,8,9,10,11,12,13,14,15);
    v16bf b0 = __builtin_shufflevector(pb0[half], pb0[2 + half],
                                       0,1,2,3,4,5,6,7,8,9,10,11,12,13,14,15);
    v16bf b1 = __builtin_shufflevector(pb1[half], pb1[2 + half],
                                       0,1,2,3,4,5,6,7,8,9,10,11,12,13,14,15);
    acc00 = __builtin_amdgcn_wmma_f32_16x16x32_bf16(false, a0, false, b0, (short)0, acc00, false, false);
    acc01 = __builtin_amdgcn_wmma_f32_16x16x32_bf16(false, a0, false, b1, (short)0, acc01, false, false);
    acc10 = __builtin_amdgcn_wmma_f32_16x16x32_bf16(false, a1, false, b0, (short)0, acc10, false, false);
    acc11 = __builtin_amdgcn_wmma_f32_16x16x32_bf16(false, a1, false, b1, (short)0, acc11, false, false);
    __syncthreads();   // protect buf from tile k+2's DMA
    buf ^= 1;
  }

  int gn0 = n0 + nw + lm;
#pragma unroll
  for (int r = 0; r < 8; ++r) {
    int gm0 = m0 + mw + r + 8 * half;
    int gm1 = gm0 + 16;
    if (gm0 < M) {
      C[(size_t)gm0 * N + gn0]      = acc00[r];
      C[(size_t)gm0 * N + gn0 + 16] = acc01[r];
    }
    if (gm1 < M) {
      C[(size_t)gm1 * N + gn0]      = acc10[r];
      C[(size_t)gm1 * N + gn0 + 16] = acc11[r];
    }
  }
}

// ---------------- gcn 1x1 over [h1;h2] + gated residual + dense ----------------
__global__ __launch_bounds__(256)
void k_gcn_combine(const float* __restrict__ h, const float* __restrict__ gw,
                   const float* __restrict__ gb, const float* __restrict__ res,
                   const float* __restrict__ dOld, float* __restrict__ dNew,
                   float* __restrict__ xo, int Tin, int Tout) {
  __shared__ float sgw[2048];
  for (int j = threadIdx.x; j < 2048; j += 256) sgw[j] = gw[j];
  __syncthreads();
  int idx = blockIdx.x * blockDim.x + threadIdx.x;
  int total = BATCH * RC * V_N * Tout;
  if (idx >= total) return;
  int l = idx % Tout;
  int v = (idx / Tout) % V_N;
  int co = (idx / (Tout * V_N)) % RC;
  int bb = idx / (Tout * V_N * RC);
  int N = BATCH * DC * Tout;
  float acc = gb[co];
  for (int ci = 0; ci < DC; ++ci) {
    int n = (bb * DC + ci) * Tout + l;
    acc += sgw[co * 64 + ci] * h[(size_t)v * N + n]
         + sgw[co * 64 + 32 + ci] * h[(size_t)(V_N + v) * N + n];
  }
  size_t ri = ((size_t)(bb * RC + co) * V_N + v) * Tin + (Tin - Tout) + l;
  float dn = res[ri] + (dOld ? dOld[ri] : 0.0f);
  float g = sigf(acc);
  size_t oi = ((size_t)(bb * RC + co) * V_N + v) * Tout + l;
  dNew[oi] = dn;
  xo[oi] = acc * g + dn * (1.0f - g);
}

// ---------------- layernorm (deterministic two-stage reduction) ----------------
__global__ void k_ln_stats(const float* __restrict__ x, float* __restrict__ part,
                           int Nel, int nblk) {
  __shared__ float r1[256], r2[256];
  int bb = blockIdx.x / nblk, blk = blockIdx.x % nblk;
  int base = blk * 4096;
  float s = 0.f, s2 = 0.f;
  for (int i = base + threadIdx.x; i < base + 4096 && i < Nel; i += 256) {
    float v = x[(size_t)bb * Nel + i];
    s += v; s2 += v * v;
  }
  r1[threadIdx.x] = s; r2[threadIdx.x] = s2;
  __syncthreads();
  for (int st = 128; st > 0; st >>= 1) {
    if (threadIdx.x < st) { r1[threadIdx.x] += r1[threadIdx.x + st]; r2[threadIdx.x] += r2[threadIdx.x + st]; }
    __syncthreads();
  }
  if (threadIdx.x == 0) {
    part[(size_t)(bb * nblk + blk) * 2]     = r1[0];
    part[(size_t)(bb * nblk + blk) * 2 + 1] = r2[0];
  }
}

__global__ void k_ln_reduce(const float* __restrict__ part, int nblk, float Nel,
                            float* __restrict__ musig) {
  __shared__ float r1[256], r2[256];
  int bb = blockIdx.x;
  float s = 0.f, s2 = 0.f;
  for (int j = threadIdx.x; j < nblk; j += 256) {
    s  += part[(size_t)(bb * nblk + j) * 2];
    s2 += part[(size_t)(bb * nblk + j) * 2 + 1];
  }
  r1[threadIdx.x] = s; r2[threadIdx.x] = s2;
  __syncthreads();
  for (int st = 128; st > 0; st >>= 1) {
    if (threadIdx.x < st) { r1[threadIdx.x] += r1[threadIdx.x + st]; r2[threadIdx.x] += r2[threadIdx.x + st]; }
    __syncthreads();
  }
  if (threadIdx.x == 0) {
    float mu = r1[0] / Nel;
    float var = r2[0] / Nel - mu * mu;
    musig[bb * 2] = mu;
    musig[bb * 2 + 1] = rsqrtf(var + 1e-5f);
  }
}

__global__ void k_ln_apply(float* __restrict__ x, const float* __restrict__ musig,
                           const float* __restrict__ lw, const float* __restrict__ lb,
                           int Nel) {
  int idx = blockIdx.x * blockDim.x + threadIdx.x;
  if (idx >= BATCH * Nel) return;
  int bb = idx / Nel, i = idx % Nel;
  x[idx] = (x[idx] - musig[bb * 2]) * musig[bb * 2 + 1] * lw[i] + lb[i];
}

// ---------------- end convs ----------------
__global__ void k_pack_wbg(const float* __restrict__ eb1w, const float* __restrict__ eg1w,
                           bf16_t* __restrict__ Wbg) {
  int idx = blockIdx.x * blockDim.x + threadIdx.x;
  if (idx >= 1024 * 1536) return;
  int r = idx / 1536, k = idx % 1536;
  Wbg[idx] = (bf16_t)((r < 512) ? eb1w[idx] : eg1w[(size_t)(r - 512) * 1536 + k]);
}

__global__ void k_bias_relu(float* __restrict__ hbg, const float* __restrict__ eb1b,
                            const float* __restrict__ eg1b) {
  int idx = blockIdx.x * blockDim.x + threadIdx.x;
  if (idx >= 1024 * NBV) return;
  int row = idx / NBV;
  float bias = (row < 512) ? eb1b[row] : eg1b[row - 512];
  hbg[idx] = fmaxf(hbg[idx] + bias, 0.0f);
}

__global__ void k_end(const float* __restrict__ hbg,
                      const float* __restrict__ eb2w, const float* __restrict__ eb2b,
                      const float* __restrict__ eg2w, const float* __restrict__ eg2b,
                      float* __restrict__ pb, float* __restrict__ pg) {
  int idx = blockIdx.x * blockDim.x + threadIdx.x;
  if (idx >= NBV * 14) return;
  int n = idx % NBV, t = idx / NBV;
  float ab = 0.f, ag = 0.f;
  for (int e = 0; e < 512; ++e) {
    ab += eb2w[t * 512 + e] * hbg[(size_t)e * NBV + n];
    ag += eg2w[t * 512 + e] * hbg[(size_t)(512 + e) * NBV + n];
  }
  pb[(size_t)t * NBV + n] = sigf(ab + eb2b[t]);
  pg[(size_t)t * NBV + n] = sigf(ag + eg2b[t]);
}

// ---------------- SIR scan ----------------
__global__ void k_sir_init(const float* __restrict__ SIR, float* __restrict__ S,
                           float* __restrict__ I, float* __restrict__ R) {
  int idx = blockIdx.x * blockDim.x + threadIdx.x;
  if (idx >= NBV) return;
  int bb = idx / V_N, v = idx % V_N;
  size_t base = ((size_t)(bb * 14 + 13) * V_N + v) * 3;
  S[idx] = SIR[base]; I[idx] = SIR[base + 1]; R[idx] = SIR[base + 2];
}

__global__ void k_sir_r(const float* __restrict__ S, const float* __restrict__ I,
                        const float* __restrict__ R, float* __restrict__ rb,
                        float* __restrict__ tot) {
  int idx = blockIdx.x * blockDim.x + threadIdx.x;
  if (idx >= NBV) return;
  float t = S[idx] + I[idx] + R[idx];
  tot[idx] = t;
  rb[idx] = I[idx] / t;
}

__global__ void k_sir_step(const float* __restrict__ mob, const float* __restrict__ mobT,
                           const float* __restrict__ rb, const float* __restrict__ tot,
                           const float* __restrict__ Sa, const float* __restrict__ Ia,
                           const float* __restrict__ Ra,
                           const float* __restrict__ pbt, const float* __restrict__ pgt,
                           float* __restrict__ Sb, float* __restrict__ Ib,
                           float* __restrict__ Rb, float* __restrict__ out, int t) {
  int idx = blockIdx.x * blockDim.x + threadIdx.x;
  if (idx >= NBV) return;
  int bb = idx / V_N, v = idx % V_N;
  const float* mtr = mobT + (size_t)v * V_N;
  const float* mr  = mob  + (size_t)v * V_N;
  const float* rr  = rb + bb * V_N;
  const float* ir  = Ia + bb * V_N;
  float t1 = 0.f, t2 = 0.f;
  for (int w = 0; w < V_N; ++w) { t1 += mtr[w] * rr[w]; t2 += mr[w] * ir[w]; }
  float prop = t1 + t2 / tot[idx];
  float In = pbt[idx] * prop;
  float rec = Ia[idx] * pgt[idx];
  Sb[idx] = Sa[idx] - In;
  Ib[idx] = Ia[idx] + In - rec;
  Rb[idx] = rec + Ra[idx];
  out[(size_t)(bb * 14 + t) * V_N + v] = In;
}

// ---------------------------------------------------------------------------
extern "C" void kernel_launch(void* const* d_in, const int* in_sizes, int n_in,
                              void* d_out, int out_size, void* d_ws, size_t ws_size,
                              hipStream_t stream) {
  const float* x_node  = (const float*)d_in[0];
  const float* SIRin   = (const float*)d_in[1];
  const float* g_res   = (const float*)d_in[2];
  const float* max_log = (const float*)d_in[3];
  const float* start_w = (const float*)d_in[4];
  const float* start_b = (const float*)d_in[5];
  const float* filter_w = (const float*)d_in[6];
  const float* filter_b = (const float*)d_in[7];
  const float* gate_w  = (const float*)d_in[8];
  const float* gate_b  = (const float*)d_in[9];
  const float* skip_w  = (const float*)d_in[10];
  const float* skip_b  = (const float*)d_in[11];
  const float* gcn_w   = (const float*)d_in[12];
  const float* gcn_b   = (const float*)d_in[13];
  const float* ln_w[NLAY]; const float* ln_b[NLAY];
  for (int i = 0; i < NLAY; ++i) { ln_w[i] = (const float*)d_in[14 + i]; ln_b[i] = (const float*)d_in[20 + i]; }
  const float* eb1_w = (const float*)d_in[26];
  const float* eb1_b = (const float*)d_in[27];
  const float* eb2_w = (const float*)d_in[28];
  const float* eb2_b = (const float*)d_in[29];
  const float* eg1_w = (const float*)d_in[30];
  const float* eg1_b = (const float*)d_in[31];
  const float* eg2_w = (const float*)d_in[32];
  const float* eg2_b = (const float*)d_in[33];
  float* out = (float*)d_out;

  // ---- workspace carve-out (256B aligned) ----
  char* base = (char*)d_ws;
  size_t off = 0;
  auto alloc = [&](size_t bytes) -> char* {
    off = (off + 255) & ~(size_t)255;
    char* p = base + off;
    off += bytes;
    return p;
  };
  float*  mob   = (float*)alloc((size_t)V_N * V_N * 4);
  float*  mobT  = (float*)alloc((size_t)V_N * V_N * 4);
  float*  rs    = (float*)alloc(4000 * 4);
  bf16_t* A12   = (bf16_t*)alloc((size_t)MP12 * KP * 2);
  float*  xA    = (float*)alloc((size_t)BATCH * RC * V_N * 15 * 4);
  float*  xB    = (float*)alloc((size_t)BATCH * RC * V_N * 15 * 4);
  float*  x1    = (float*)alloc((size_t)BATCH * DC * V_N * 14 * 4);
  bf16_t* Xt    = (bf16_t*)alloc((size_t)(BATCH * DC * 14) * KP * 2);  // X^T (N x KP)
  float*  hbuf  = (float*)alloc((size_t)1024 * NBV * 4);   // also holds 4000 x 3584
  float*  dA    = (float*)alloc((size_t)BATCH * RC * V_N * 14 * 4);
  float*  dB    = (float*)alloc((size_t)BATCH * RC * V_N * 14 * 4);
  bf16_t* feat  = (bf16_t*)alloc((size_t)NBV * 1536 * 2);  // feat^T (NBV x 1536)
  bf16_t* Wbg   = (bf16_t*)alloc((size_t)1024 * 1536 * 2);
  float*  part  = (float*)alloc((size_t)BATCH * 256 * 2 * 4);
  float*  musig = (float*)alloc(BATCH * 2 * 4);
  float*  pb    = (float*)alloc((size_t)14 * NBV * 4);
  float*  pg    = (float*)alloc((size_t)14 * NBV * 4);
  float*  Sst[2], *Ist[2], *Rst[2];
  for (int j = 0; j < 2; ++j) {
    Sst[j] = (float*)alloc(NBV * 4);
    Ist[j] = (float*)alloc(NBV * 4);
    Rst[j] = (float*)alloc(NBV * 4);
  }
  float* rbuf = (float*)alloc(NBV * 4);
  float* tot  = (float*)alloc(NBV * 4);
  (void)ws_size; (void)in_sizes; (void)n_in; (void)out_size;

  // ---- adjacency ----
  k_mob_build<<<CDIV(V_N * V_N, 256), 256, 0, stream>>>(g_res, max_log, mob, mobT);
  k_rowsum<<<4000, 256, 0, stream>>>(mob, mobT, rs);
  k_a12<<<CDIV(MP12 * KP, 256), 256, 0, stream>>>(mob, mobT, rs, A12);
  k_pack_wbg<<<CDIV(1024 * 1536, 256), 256, 0, stream>>>(eb1_w, eg1_w, Wbg);

  // ---- start conv ----
  k_start<<<CDIV(BATCH * RC * V_N * 15, 256), 256, 0, stream>>>(x_node, start_w, start_b, xA);

  // ---- wavenet layers ----
  const int Tin[NLAY]  = {15, 14, 12, 8, 7, 5};
  const int Tout[NLAY] = {14, 12, 8, 7, 5, 1};
  const int dil[NLAY]  = {1, 2, 4, 1, 2, 4};
  float *cx = xA, *nx = xB, *dOld = nullptr, *dNew = dA;

  for (int i = 0; i < NLAY; ++i) {
    int Ti = Tin[i], To = Tout[i];
    int N = BATCH * DC * To;
    k_filtgate<<<CDIV(BATCH * DC * V_N * To, 256), 256, 0, stream>>>(
        cx, filter_w + i * 2048, filter_b + i * 32, gate_w + i * 2048, gate_b + i * 32,
        x1, Ti, To, dil[i]);
    k_skip<<<CDIV(BATCH * SC * V_N, 256), 256, 0, stream>>>(
        x1, skip_w + i * SC * DC, skip_b + i * SC, feat, To, (NLAY - 1 - i) * SC);
    k_pack_x<<<CDIV(N * KP, 256), 256, 0, stream>>>(x1, Xt, To);
    {
      dim3 grid(CDIV(2 * V_N, 128), N / 64);
      k_gemm_bf16<<<grid, 256, 0, stream>>>(A12, Xt, hbuf, 2 * V_N, N, KP);
    }
    k_gcn_combine<<<CDIV(BATCH * RC * V_N * To, 256), 256, 0, stream>>>(
        hbuf, gcn_w + i * 2048, gcn_b + i * 32, cx, dOld, dNew, nx, Ti, To);
    int Nel = RC * V_N * To;
    int nblk = CDIV(Nel, 4096);
    k_ln_stats<<<BATCH * nblk, 256, 0, stream>>>(nx, part, Nel, nblk);
    k_ln_reduce<<<BATCH, 256, 0, stream>>>(part, nblk, (float)Nel, musig);
    k_ln_apply<<<CDIV(BATCH * Nel, 256), 256, 0, stream>>>(nx, musig, ln_w[i], ln_b[i], Nel);
    // rotate buffers
    float* t0 = cx; cx = nx; nx = t0;
    dOld = dNew; dNew = (dNew == dA) ? dB : dA;
  }

  // ---- end convs: [eb1;eg1] GEMM then 512->14 ----
  {
    dim3 grid(CDIV(1024, 128), NBV / 64);
    k_gemm_bf16<<<grid, 256, 0, stream>>>(Wbg, feat, hbuf, 1024, NBV, 1536);
  }
  k_bias_relu<<<CDIV(1024 * NBV, 256), 256, 0, stream>>>(hbuf, eb1_b, eg1_b);
  k_end<<<CDIV(NBV * 14, 256), 256, 0, stream>>>(hbuf, eb2_w, eb2_b, eg2_w, eg2_b, pb, pg);

  // ---- SIR scan ----
  k_sir_init<<<CDIV(NBV, 256), 256, 0, stream>>>(SIRin, Sst[0], Ist[0], Rst[0]);
  int cur = 0;
  for (int t = 0; t < 14; ++t) {
    int nxt = cur ^ 1;
    k_sir_r<<<CDIV(NBV, 256), 256, 0, stream>>>(Sst[cur], Ist[cur], Rst[cur], rbuf, tot);
    k_sir_step<<<CDIV(NBV, 256), 256, 0, stream>>>(
        mob, mobT, rbuf, tot, Sst[cur], Ist[cur], Rst[cur],
        pb + (size_t)t * NBV, pg + (size_t)t * NBV,
        Sst[nxt], Ist[nxt], Rst[nxt], out, t);
    cur = nxt;
  }
}
